// cfmf_46196668236068
// MI455X (gfx1250) — compile-verified, matrix operand order checked by reading, not run on hardware
//
#include <hip/hip_runtime.h>
#include <hip/hip_bf16.h>

typedef __attribute__((ext_vector_type(16))) __bf16       bf16x16;
typedef __attribute__((ext_vector_type(8)))  float        v8f;
typedef __attribute__((ext_vector_type(8)))  unsigned int u32x8;
typedef __attribute__((ext_vector_type(4)))  unsigned int u32x4;
typedef __attribute__((ext_vector_type(8)))  int          i32x8;
typedef __attribute__((ext_vector_type(4)))  int          i32x4;

#define EPSS 1e-6f

// ---- WMMA helpers (V_WMMA_F32_16X16X32_BF16, fp32 accumulate) -------------

__device__ __forceinline__ v8f wmma_bf16(bf16x16 a, bf16x16 b, v8f c) {
  return __builtin_amdgcn_wmma_f32_16x16x32_bf16(false, a, false, b, (short)0, c,
                                                 false, false);
}

// ISA 7.12.2: 16-bit operand, element j of a lane maps to
//   K = 16*(j/8) + (j%8) + 8*(lane/16)   -> fk(2t,hi) is even, pairs contiguous
__device__ __forceinline__ int fk(int j, int hi) {
  return ((j >> 3) << 4) + (j & 7) + (hi << 3);
}

// A-fragment from bf16 row-major [row][K] (ld even, k0 mult of 32): 8 dword loads
__device__ __forceinline__ bf16x16 ldA_pk(const __bf16* s, int row0, int ld, int k0,
                                          int lane) {
  const unsigned int* p = (const unsigned int*)s;
  int hi = lane >> 4, r = row0 + (lane & 15);
  u32x8 w;
#pragma unroll
  for (int t = 0; t < 8; ++t) w[t] = p[(r * ld + k0 + fk(2 * t, hi)) >> 1];
  return __builtin_bit_cast(bf16x16, w);
}

// B-fragment from pair-interleaved u32 layout: slot[(k>>1)*ncols + n] = {bf16(k,n), bf16(k+1,n)}
__device__ __forceinline__ bf16x16 ldB_pk(const unsigned int* s, int k0, int ncols,
                                          int n0, int lane) {
  int hi = lane >> 4, n = n0 + (lane & 15);
  u32x8 w;
#pragma unroll
  for (int t = 0; t < 8; ++t) w[t] = s[((k0 + fk(2 * t, hi)) >> 1) * ncols + n];
  return __builtin_bit_cast(bf16x16, w);
}

__device__ __forceinline__ bf16x16 neg_bf16(bf16x16 a) {
  u32x8 w = __builtin_bit_cast(u32x8, a);
#pragma unroll
  for (int t = 0; t < 8; ++t) w[t] ^= 0x80008000u;
  return __builtin_bit_cast(bf16x16, w);
}

// ---- TDM: DMA a [nrows x ncols] f32 tile (full-width, contiguous rows) to LDS
#if __has_builtin(__builtin_amdgcn_tensor_load_to_lds) && \
    __has_builtin(__builtin_amdgcn_s_wait_tensorcnt)
#define HAVE_TDM 1
__device__ __forceinline__ void tdm_load_rows_f32(const float* gbase, unsigned lds_off,
                                                  unsigned row0, unsigned nrows,
                                                  unsigned ncols, unsigned tot_rows) {
  unsigned long long ga =
      (unsigned long long)(const void*)(gbase + (size_t)row0 * ncols);
  u32x4 g0;
  g0[0] = 1u;                                            // count=1, user descriptor
  g0[1] = lds_off;                                       // lds_addr (bytes)
  g0[2] = (unsigned)(ga & 0xffffffffu);                  // global_addr[31:0]
  g0[3] = (unsigned)((ga >> 32) & 0x01ffffffu) | (2u << 30);  // addr[56:32] | type=2
  i32x8 g1;
  g1[0] = 0x00020000;              // wg_mask=0, data_size=2 (4B), no pad/iterate
  g1[1] = (int)(ncols << 16);      // tensor_dim0[15:0]      (bits 63:48)
  g1[2] = (int)(tot_rows << 16);   // tensor_dim1[15:0]      (bits 95:80)
  g1[3] = (int)(ncols << 16);      // tile_dim0              (bits 127:112)
  g1[4] = (int)nrows;              // tile_dim1              (bits 143:128)
  g1[5] = (int)ncols;              // tensor_dim0_stride[31:0]
  g1[6] = 0;
  g1[7] = 0;
  i32x4 z4 = {0, 0, 0, 0};
#if defined(__clang_major__) && __clang_major__ >= 23
  i32x8 z8 = {0, 0, 0, 0, 0, 0, 0, 0};
  __builtin_amdgcn_tensor_load_to_lds(g0, g1, z4, z4, z8, 0);
#else
  __builtin_amdgcn_tensor_load_to_lds(g0, g1, z4, z4, 0);
#endif
}
#else
#define HAVE_TDM 0
#endif

// ---- Kernel 1: AA = A^H A -> bf16 pair-interleaved [q/2][p] layouts; diag f32
__global__ __launch_bounds__(256) void k_prep(const float* __restrict__ Ar,
                                              const float* __restrict__ Ai,
                                              unsigned int* AArTp, unsigned int* AAiTp,
                                              float* Dre, unsigned int* maxbits) {
  int idx = blockIdx.x * 256 + threadIdx.x;   // 65536 = 256*256
  int q = idx >> 8, p = idx & 255;
  float sr = 0.f, si = 0.f;
  for (int n = 0; n < 128; ++n) {
    float arp = Ar[n * 256 + p], aip = Ai[n * 256 + p];
    float arq = Ar[n * 256 + q], aiq = Ai[n * 256 + q];
    sr += arp * arq + aip * aiq;          // real AA[p][q]
    si += arp * aiq - aip * arq;          // imag AA[p][q]
  }
  // B-frag element (k=q, n=p): bf16 slot 2*((q>>1)*256+p) + (q&1)
  int e = 2 * ((q >> 1) * 256 + p) + (q & 1);
  ((__bf16*)AArTp)[e] = (__bf16)sr;
  ((__bf16*)AAiTp)[e] = (__bf16)si;
  if (p == q) Dre[p] = sr;
  if (idx == 0) *maxbits = 0u;
}

// ---- Kernel 2: bf16 casts: Y row-major; A / A^T in pair-interleaved B layouts
__global__ __launch_bounds__(256) void k_cast(const float* __restrict__ Yr,
                                              const float* __restrict__ Yi,
                                              const float* __restrict__ Ar,
                                              const float* __restrict__ Ai,
                                              __bf16* Yr_h, __bf16* Yi_h,
                                              unsigned int* Arp, unsigned int* Aip,
                                              unsigned int* ArTp, unsigned int* AiTp) {
  int idx = blockIdx.x * 256 + threadIdx.x;   // 262144
  if (idx < 32768) {
    float ar = Ar[idx], ai = Ai[idx];
    int n = idx >> 8, m = idx & 255;
    int ea = 2 * ((n >> 1) * 256 + m) + (n & 1);   // B layout (k=n, col=m), N=256
    ((__bf16*)Arp)[ea] = (__bf16)ar;
    ((__bf16*)Aip)[ea] = (__bf16)ai;
    int et = 2 * ((m >> 1) * 128 + n) + (m & 1);   // B layout (k=m, col=n), N=128
    ((__bf16*)ArTp)[et] = (__bf16)ar;
    ((__bf16*)AiTp)[et] = (__bf16)ai;
  }
  Yr_h[idx] = (__bf16)Yr[idx];
  Yi_h[idx] = (__bf16)Yi[idx];
}

// ---- Kernel 3: AY = A^H Y  (complex GEMM [2048,128]x[128,256] via WMMA) ---
__global__ __launch_bounds__(32) void k_ay(const __bf16* __restrict__ Yr_h,
                                           const __bf16* __restrict__ Yi_h,
                                           const unsigned int* __restrict__ Arp,
                                           const unsigned int* __restrict__ Aip,
                                           float* __restrict__ AYr,
                                           float* __restrict__ AYi) {
  int lane = threadIdx.x;
  int m0 = blockIdx.x * 16, b0 = blockIdx.y * 16;
  v8f aR = {}, aI = {};
  for (int k = 0; k < 128; k += 32) {
    bf16x16 yr  = ldA_pk(Yr_h, b0, 128, k, lane);
    bf16x16 yi  = ldA_pk(Yi_h, b0, 128, k, lane);
    bf16x16 yrn = neg_bf16(yr);
    bf16x16 br  = ldB_pk(Arp, k, 256, m0, lane);
    bf16x16 bi  = ldB_pk(Aip, k, 256, m0, lane);
    aR = wmma_bf16(yr,  br, aR);   // AYr = Yr*Ar + Yi*Ai   (conj(A))
    aR = wmma_bf16(yi,  bi, aR);
    aI = wmma_bf16(yi,  br, aI);   // AYi = Yi*Ar - Yr*Ai
    aI = wmma_bf16(yrn, bi, aI);
  }
  int hi = lane >> 4, lq = lane & 15;
#pragma unroll
  for (int r = 0; r < 8; ++r) {
    int b = b0 + r + 8 * hi;
    AYr[b * 256 + m0 + lq] = aR[r];
    AYi[b * 256 + m0 + lq] = aI[r];
  }
}

// ---- Kernel 4: fused conv1(2->32) + conv2(32->64, implicit-GEMM WMMA) + mean over M
__global__ __launch_bounds__(128) void k_hyper(const float* __restrict__ AYr,
                                               const float* __restrict__ AYi,
                                               const float* __restrict__ c1w,
                                               const float* __restrict__ c1b,
                                               const float* __restrict__ c2w,
                                               const float* __restrict__ c2b,
                                               float* __restrict__ pooled) {
  __shared__ __attribute__((aligned(4))) __bf16 W2h[64 * 288]; // [co][ci*9+kh*3+kw]
  __shared__ float W1[576];
  __shared__ float B1[32], B2[64];
  __shared__ float F1[3 * 32 * 18];   // rolling conv1 rows: [slot][ci][col(18)]
  int tid = threadIdx.x, lane = tid & 31, wv = tid >> 5;
  int hi = lane >> 4, lq = lane & 15;
  int wbase = blockIdx.x * 16;

  for (int i = tid; i < 64 * 288; i += 128) W2h[i] = (__bf16)c2w[i];
  for (int i = tid; i < 576; i += 128) W1[i] = c1w[i];
  if (tid < 32) B1[tid] = c1b[tid];
  if (tid < 64) B2[tid] = c2b[tid];
  for (int i = tid; i < 576; i += 128) F1[2 * 576 + i] = 0.f;  // row -1 -> slot 2

  auto conv1row = [&](int hr, int slot) {
    for (int i = tid; i < 576; i += 128) {
      int co = i / 18, col = i % 18;
      int w = wbase - 1 + col;
      float v = 0.f;
      if (w >= 0 && w < 2048) {
        v = B1[co];
        for (int cin = 0; cin < 2; ++cin) {
          const float* X = cin ? AYi : AYr;   // X[c][h][w] = AY_c[w*256 + h]
          for (int dh = 0; dh < 3; ++dh) {
            int hh = hr + dh - 1;
            if (hh < 0 || hh >= 256) continue;
            for (int dw = 0; dw < 3; ++dw) {
              int ww = w + dw - 1;
              if (ww < 0 || ww >= 2048) continue;
              v += W1[co * 18 + cin * 9 + dh * 3 + dw] * X[ww * 256 + hh];
            }
          }
        }
        v = v > 0.f ? v : 0.f;
      }
      F1[slot * 576 + i] = v;
    }
  };
  conv1row(0, 0);
  __syncthreads();

  int cobase = wv * 16;
  float acc[8] = {0.f, 0.f, 0.f, 0.f, 0.f, 0.f, 0.f, 0.f};
  for (int h = 0; h < 256; ++h) {
    int hn = h + 1, sl = hn % 3;
    if (hn < 256) conv1row(hn, sl);
    else for (int i = tid; i < 576; i += 128) F1[sl * 576 + i] = 0.f;
    __syncthreads();
    // conv2 row h: out[16co x 16w] = W2[16co x 288] @ patch[288 x 16w], K = 9x32
    v8f cf = {};
    for (int ks = 0; ks < 288; ks += 32) {
      bf16x16 af = ldA_pk(W2h, cobase, 288, ks, lane);
      bf16x16 bp;
#pragma unroll
      for (int j = 0; j < 16; ++j) {
        int k = ks + fk(j, hi);
        int ci = k / 9, r9 = k - ci * 9, dh = r9 / 3, dw = r9 - dh * 3;
        int hh = h + dh - 1;
        float v = 0.f;
        if (hh >= 0 && hh < 256) v = F1[(hh % 3) * 576 + ci * 18 + lq + dw];
        bp[j] = (__bf16)v;
      }
      cf = wmma_bf16(af, bp, cf);
    }
#pragma unroll
    for (int r = 0; r < 8; ++r) {
      float v = cf[r] + B2[cobase + r + 8 * hi];
      acc[r] += v > 0.f ? v : 0.f;
    }
    __syncthreads();
  }
#pragma unroll
  for (int r = 0; r < 8; ++r) {
    int co = cobase + r + 8 * hi;
    pooled[co * 2048 + wbase + lq] = acc[r] * (1.f / 256.f);
  }
}

// ---- Kernel 5: conv1d heads -> c[B,9], d[B,9], T[B,10]; last-row diagnostics
__global__ __launch_bounds__(32) void k_heads(const float* __restrict__ pooled,
                                              const float* hcw, const float* hcb,
                                              const float* hdw, const float* hdb,
                                              const float* hTw, const float* hTb,
                                              float* carr, float* darr, float* tarr,
                                              float* dout) {
  int b = blockIdx.x, j = threadIdx.x;
  if (j >= 28) return;
  const float* w; const float* bias; int jj, which;
  if (j < 9)       { w = hcw + j * 192;        bias = hcb; jj = j;      which = 0; }
  else if (j < 18) { w = hdw + (j - 9) * 192;  bias = hdb; jj = j - 9;  which = 1; }
  else             { w = hTw + (j - 18) * 192; bias = hTb; jj = j - 18; which = 2; }
  float s = bias[jj];
  for (int ci = 0; ci < 64; ++ci)
    for (int dw = 0; dw < 3; ++dw) {
      int bb = b + dw - 1;
      if (bb < 0 || bb >= 2048) continue;
      s += w[ci * 3 + dw] * pooled[ci * 2048 + bb];
    }
  float a = fabsf(s), outv;
  if (which == 0)      { outv = fminf(fmaxf(a, 1e-6f), 100.f); carr[b * 9 + jj]  = outv; }
  else if (which == 1) { outv = fmaxf(a, 1e-6f);               darr[b * 9 + jj]  = outv; }
  else                 { outv = fminf(fmaxf(a, 1e-6f), 100.f); tarr[b * 10 + jj] = outv; }
  if (b == 2047) dout[524288 + (which == 0 ? jj : which == 1 ? 9 + jj : 18 + jj)] = outv;
}

// ---- Kernel 6: persistent 10-layer SBL/VAMP loop; 16 batches per block ----
__global__ __launch_bounds__(256) void k_iter(const float* __restrict__ AYr_g,
                                              const float* __restrict__ AYi_g,
                                              const float* __restrict__ Yr_g,
                                              const float* __restrict__ Yi_g,
                                              const unsigned int* __restrict__ AArTp,
                                              const unsigned int* __restrict__ AAiTp,
                                              const unsigned int* __restrict__ ArTp,
                                              const unsigned int* __restrict__ AiTp,
                                              const float* __restrict__ Dre,
                                              const float* __restrict__ carr,
                                              const float* __restrict__ darr,
                                              const float* __restrict__ tarr,
                                              const float* a0p, const float* b0p,
                                              float* __restrict__ Uabs) {
  __shared__ __attribute__((aligned(4))) __bf16 Urh[16 * 256];  // U state in bf16
  __shared__ __attribute__((aligned(4))) __bf16 Uih[16 * 256];  // (= WMMA precision)
  __shared__ float Lam_s[16 * 256];
  __shared__ float Yr_s[16 * 128], Yi_s[16 * 128];   // TDM-staged Y tile
  __shared__ float D_s[256];
  __shared__ float eps_s[16], epso_s[16], e1_s[16];
  __shared__ float c_s[16 * 9], d_s[16 * 9], T_s[16 * 10];
  int tid = threadIdx.x, lane = tid & 31, wv = tid >> 5;
  int hi = lane >> 4, lq = lane & 15;
  int bbase = blockIdx.x * 16;
  float b0v = b0p[0];
  float a_val = a0p[0] + 128.0f;   // a = a0 + Nmeas

  for (int i = tid; i < 16 * 256; i += 256) {
    Urh[i] = (__bf16)0.f; Uih[i] = (__bf16)0.f;
  }
  D_s[tid] = Dre[tid];
  for (int i = tid; i < 16 * 9; i += 256) {
    int b = i / 9, j = i % 9;
    c_s[i] = carr[(bbase + b) * 9 + j];
    d_s[i] = darr[(bbase + b) * 9 + j];
  }
  for (int i = tid; i < 16 * 10; i += 256) {
    int b = i / 10, j = i % 10;
    T_s[i] = tarr[(bbase + b) * 10 + j];
  }
  if (tid < 16) eps_s[tid] = a0p[0] / b0v;

#if HAVE_TDM
  if (tid == 0) {   // wave0 issues the tensor DMA; others branch around (EXECZ)
    tdm_load_rows_f32(Yr_g, (unsigned)(unsigned long long)(const void*)Yr_s,
                      (unsigned)bbase, 16u, 128u, 2048u);
    tdm_load_rows_f32(Yi_g, (unsigned)(unsigned long long)(const void*)Yi_s,
                      (unsigned)bbase, 16u, 128u, 2048u);
    __builtin_amdgcn_s_wait_tensorcnt(0);
  }
#else
  for (int i = tid; i < 16 * 128; i += 256) {
    Yr_s[i] = Yr_g[bbase * 128 + i];
    Yi_s[i] = Yi_g[bbase * 128 + i];
  }
#endif
  __syncthreads();
  {
    int b = tid >> 4;
    float l0 = c_s[b * 9] / d_s[b * 9];
    for (int i = 0; i < 16; ++i) Lam_s[b * 256 + (tid & 15) * 16 + i] = l0;
  }
  __syncthreads();

  __builtin_prefetch(AYr_g + (size_t)bbase * 256, 0, 1);
  __builtin_prefetch(AYi_g + (size_t)bbase * 256, 0, 1);

  for (int k = 0; k < 10; ++k) {
    // (A) V = AA @ U  (complex WMMA; each wave owns p-tiles {wv, wv+8})
    v8f VR[2], VI[2];
#pragma unroll
    for (int t = 0; t < 2; ++t) {
      int p0 = (wv + 8 * t) * 16;
      v8f cr = {}, ci = {};
      for (int kk = 0; kk < 256; kk += 32) {
        bf16x16 ur  = ldA_pk(Urh, 0, 256, kk, lane);
        bf16x16 ui  = ldA_pk(Uih, 0, 256, kk, lane);
        bf16x16 uin = neg_bf16(ui);
        bf16x16 br  = ldB_pk(AArTp, kk, 256, p0, lane);
        bf16x16 bi  = ldB_pk(AAiTp, kk, 256, p0, lane);
        cr = wmma_bf16(ur,  br, cr);   // Re: Ur*AAr - Ui*AAi
        cr = wmma_bf16(uin, bi, cr);
        ci = wmma_bf16(ur,  bi, ci);   // Im: Ur*AAi + Ui*AAr
        ci = wmma_bf16(ui,  br, ci);
      }
      VR[t] = cr; VI[t] = ci;
    }
    __syncthreads();
    // (B) U = eps*(Tk*U - V + AY)/(eps*Tk + Lam + eps_s)
#pragma unroll
    for (int t = 0; t < 2; ++t) {
      int p = (wv + 8 * t) * 16 + lq;
#pragma unroll
      for (int r = 0; r < 8; ++r) {
        int b = r + 8 * hi, gb = bbase + b;
        float epsb = eps_s[b];
        float Tk = T_s[b * 10 + k];
        float lam = Lam_s[b * 256 + p];
        float ayr = AYr_g[gb * 256 + p], ayi = AYi_g[gb * 256 + p];
        float den = epsb * Tk + lam + EPSS;
        float uor = (float)Urh[b * 256 + p], uoi = (float)Uih[b * 256 + p];
        float unr = epsb * (Tk * uor - VR[t][r] + ayr) / den;
        float uni = epsb * (Tk * uoi - VI[t][r] + ayi) / den;
        Urh[b * 256 + p] = (__bf16)unr;
        Uih[b * 256 + p] = (__bf16)uni;
        if (k == 9) Uabs[gb * 256 + p] = sqrtf(unr * unr + uni * uni);
      }
    }
    if (k == 9) break;
    if (tid < 16) { e1_s[tid] = 0.f; epso_s[tid] = eps_s[tid]; }
    __syncthreads();
    // (C) r = Y - A U ; eps1 = ||r||^2 per batch (each wave owns one n-tile)
    {
      int n0 = wv * 16;
      v8f cr = {}, ci = {};
      for (int kk = 0; kk < 256; kk += 32) {
        bf16x16 ur  = ldA_pk(Urh, 0, 256, kk, lane);
        bf16x16 ui  = ldA_pk(Uih, 0, 256, kk, lane);
        bf16x16 uin = neg_bf16(ui);
        bf16x16 br  = ldB_pk(ArTp, kk, 128, n0, lane);
        bf16x16 bi  = ldB_pk(AiTp, kk, 128, n0, lane);
        cr = wmma_bf16(ur,  br, cr);   // Re(AU) = Ar*Ur - Ai*Ui
        cr = wmma_bf16(uin, bi, cr);
        ci = wmma_bf16(ur,  bi, ci);   // Im(AU) = Ar*Ui + Ai*Ur
        ci = wmma_bf16(ui,  br, ci);
      }
#pragma unroll
      for (int r = 0; r < 8; ++r) {
        int b = r + 8 * hi, n = n0 + lq;
        float rr = Yr_s[b * 128 + n] - cr[r];
        float ri = Yi_s[b * 128 + n] - ci[r];
        atomicAdd(&e1_s[b], rr * rr + ri * ri);
      }
    }
    __syncthreads();
    // (D) eps2 = sum D*Sigma (Sigma from old eps/Lam); eps update
    if (tid < 16) {
      int b = tid;
      float eo = epso_s[b], s = 0.f;
      for (int m = 0; m < 256; ++m) {
        float dd = D_s[m];
        s += dd / (eo * dd + Lam_s[b * 256 + m] + EPSS);
      }
      eps_s[b] = a_val / (b0v + e1_s[b] + s + EPSS);
    }
    __syncthreads();
    // (E) Lam = (c_k+1)/(d_k + |U|^2 + Sigma + eps_s)
    {
      int b = tid >> 4;
      float eo = epso_s[b];
      float ck = c_s[b * 9 + k], dk = d_s[b * 9 + k];
      for (int i = 0; i < 16; ++i) {
        int m = (tid & 15) * 16 + i;
        float lamold = Lam_s[b * 256 + m];
        float sig = 1.f / (eo * D_s[m] + lamold + EPSS);
        float urv = (float)Urh[b * 256 + m], uiv = (float)Uih[b * 256 + m];
        Lam_s[b * 256 + m] = (ck + 1.f) / (dk + urv * urv + uiv * uiv + sig + EPSS);
      }
    }
    __syncthreads();
  }
}

// ---- Kernel 7/8: global max of |U| then normalize into d_out --------------
__global__ __launch_bounds__(256) void k_max(const float* __restrict__ Uabs,
                                             unsigned int* maxbits) {
  __shared__ float red[256];
  int idx = blockIdx.x * 256 + threadIdx.x;
  float m = 0.f;
  for (int i = idx; i < 524288; i += gridDim.x * 256) m = fmaxf(m, Uabs[i]);
  red[threadIdx.x] = m;
  __syncthreads();
  for (int s = 128; s > 0; s >>= 1) {
    if (threadIdx.x < s) red[threadIdx.x] = fmaxf(red[threadIdx.x], red[threadIdx.x + s]);
    __syncthreads();
  }
  if (threadIdx.x == 0) atomicMax(maxbits, __float_as_uint(red[0]));  // all >= 0
}

__global__ __launch_bounds__(256) void k_norm(const float* __restrict__ Uabs,
                                              const unsigned int* maxbits,
                                              float* __restrict__ dout) {
  int i = blockIdx.x * 256 + threadIdx.x;
  float mx = __uint_as_float(*maxbits) + 1e-8f;
  if (i < 524288) dout[i] = Uabs[i] / mx;
}

// ---- host ------------------------------------------------------------------
extern "C" void kernel_launch(void* const* d_in, const int* in_sizes, int n_in,
                              void* d_out, int out_size, void* d_ws, size_t ws_size,
                              hipStream_t stream) {
  const float* Yr  = (const float*)d_in[0];
  const float* Yi  = (const float*)d_in[1];
  const float* Ar  = (const float*)d_in[2];
  const float* Ai  = (const float*)d_in[3];
  const float* c1w = (const float*)d_in[4];
  const float* c1b = (const float*)d_in[5];
  const float* c2w = (const float*)d_in[6];
  const float* c2b = (const float*)d_in[7];
  const float* hcw = (const float*)d_in[8];
  const float* hcb = (const float*)d_in[9];
  const float* hdw = (const float*)d_in[10];
  const float* hdb = (const float*)d_in[11];
  const float* hTw = (const float*)d_in[12];
  const float* hTb = (const float*)d_in[13];
  const float* a0  = (const float*)d_in[14];
  const float* b0  = (const float*)d_in[15];
  float* dout = (float*)d_out;

  char* ws = (char*)d_ws;
  size_t off = 0;
  auto take = [&](size_t bytes) -> char* {
    char* p = ws + off;
    off = (off + bytes + 255) & ~(size_t)255;
    return p;
  };
  float* AYr    = (float*)take(524288 * 4);
  float* AYi    = (float*)take(524288 * 4);
  float* UabsW  = (float*)take(524288 * 4);
  float* pooled = (float*)take(131072 * 4);
  float* carr   = (float*)take(18432 * 4);
  float* darr   = (float*)take(18432 * 4);
  float* tarr   = (float*)take(20480 * 4);
  float* Dre    = (float*)take(256 * 4);
  unsigned int* maxbits = (unsigned int*)take(4);
  unsigned int* AArTp = (unsigned int*)take(32768 * 4);  // [q/2][p] u32 pairs
  unsigned int* AAiTp = (unsigned int*)take(32768 * 4);
  unsigned int* Arp   = (unsigned int*)take(16384 * 4);  // [n/2][m]
  unsigned int* Aip   = (unsigned int*)take(16384 * 4);
  unsigned int* ArTp  = (unsigned int*)take(16384 * 4);  // [m/2][n]
  unsigned int* AiTp  = (unsigned int*)take(16384 * 4);
  __bf16* Yr_h = (__bf16*)take(262144 * 2);
  __bf16* Yi_h = (__bf16*)take(262144 * 2);

  k_prep<<<256, 256, 0, stream>>>(Ar, Ai, AArTp, AAiTp, Dre, maxbits);
  k_cast<<<1024, 256, 0, stream>>>(Yr, Yi, Ar, Ai, Yr_h, Yi_h, Arp, Aip, ArTp, AiTp);
  k_ay<<<dim3(16, 128), 32, 0, stream>>>(Yr_h, Yi_h, Arp, Aip, AYr, AYi);
  k_hyper<<<128, 128, 0, stream>>>(AYr, AYi, c1w, c1b, c2w, c2b, pooled);
  k_heads<<<2048, 32, 0, stream>>>(pooled, hcw, hcb, hdw, hdb, hTw, hTb,
                                   carr, darr, tarr, dout);
  k_iter<<<128, 256, 0, stream>>>(AYr, AYi, Yr, Yi, AArTp, AAiTp, ArTp, AiTp,
                                  Dre, carr, darr, tarr, a0, b0, UabsW);
  k_max<<<256, 256, 0, stream>>>(UabsW, maxbits);
  k_norm<<<2048, 256, 0, stream>>>(UabsW, maxbits, dout);
}
